// EulerAttention_75892072121094
// MI455X (gfx1250) — matrix operand order repaired and checked by reading, not compiled, not run on hardware
//
#include <hip/hip_runtime.h>
#include <hip/hip_bf16.h>

typedef __attribute__((ext_vector_type(16))) __bf16 v16bf;
typedef __attribute__((ext_vector_type(8)))  float  v8f;

#define PHI_F      1.6180339887498949f
#define LUT_K      651.8986469044033f     // 4096 / (2*pi)
#define LUT_STEP   0.0015339807878856412f // (2*pi) / 4096
#define INV_SCALE  0.08838834764831845f   // 1/sqrt(2*dh) = 1/sqrt(128)

// Reference LUT == sin/cos of theta snapped to a 2pi/4096 grid (nearest,
// round-half-even like jnp.round -> rintf). Periodicity removes the mod.
__device__ __forceinline__ float2 lut_sincos(float theta) {
    float ang = rintf(theta * LUT_K) * LUT_STEP;
    float s, c;
    __sincosf(ang, &s, &c);
    return make_float2(s, c);
}

__device__ __forceinline__ unsigned pack_bf16(float a, float b) {
    unsigned short ua = __builtin_bit_cast(unsigned short, (__bf16)a);
    unsigned short ub = __builtin_bit_cast(unsigned short, (__bf16)b);
    return ((unsigned)ub << 16) | ua;
}

// ---------------------------------------------------------------------------
// Kernel 1: scores[b,h,s] for a 16-wide s-tile, one wave per block.
// First streaming pass over cached_states (float4 loads), phase-encode to a
// bf16 16x128 key tile in LDS, contract vs the replicated 128-d query with
// 4x v_wmma_f32_16x16x32_bf16. Per-head 1/(1+|w_key|) staged once per block.
// ---------------------------------------------------------------------------
__global__ __launch_bounds__(32)
void euler_scores_kernel(const float* __restrict__ x,
                         const float* __restrict__ cached,
                         const float* __restrict__ t,
                         const float* __restrict__ wq, const float* __restrict__ bq,
                         const float* __restrict__ wk, const float* __restrict__ bk,
                         float* __restrict__ scores)
{
    const int s0   = blockIdx.x * 16;
    const int h    = blockIdx.y;
    const int b    = blockIdx.z;
    const int lane = threadIdx.x;

    __shared__ __bf16 qs[128];        // [cos(0..63), sin(0..63)]
    __shared__ __bf16 ks[16 * 128];   // 16 s-rows x [cos, sin]
    __shared__ float  invk[64];       // 1/(1+|w_key[h,:]|)
    __shared__ float  bkk[64];        // b_key[h,:]

    const float tphi = t[b] * PHI_F;

    // Stage key params + compute query (2 feature dims per lane).
    {
        const int d = lane * 2;
        invk[d]     = 1.0f / (1.0f + fabsf(wk[h * 64 + d]));
        invk[d + 1] = 1.0f / (1.0f + fabsf(wk[h * 64 + d + 1]));
        bkk[d]      = bk[h * 64 + d];
        bkk[d + 1]  = bk[h * 64 + d + 1];

        float iv0 = 1.0f / (1.0f + fabsf(wq[h * 64 + d]));
        float iv1 = 1.0f / (1.0f + fabsf(wq[h * 64 + d + 1]));
        float2 q0 = lut_sincos(fmaf(x[b * 1024 + h * 64 + d],     iv0, bq[h * 64 + d]     + tphi));
        float2 q1 = lut_sincos(fmaf(x[b * 1024 + h * 64 + d + 1], iv1, bq[h * 64 + d + 1] + tphi));
        *(unsigned*)&qs[d]      = pack_bf16(q0.y, q1.y);  // cos
        *(unsigned*)&qs[64 + d] = pack_bf16(q0.x, q1.x);  // sin
    }
    __syncthreads();

    // Key tile: row = lane>>1, 32 contiguous columns per lane via 8x float4.
    {
        const int row = lane >> 1;
        const int c0  = (lane & 1) * 32;
        const float4* src = (const float4*)(cached
            + ((size_t)(b * 4096 + s0 + row)) * 1024 + h * 64 + c0);
        unsigned* kcos = (unsigned*)&ks[row * 128 + c0];
        unsigned* ksin = (unsigned*)&ks[row * 128 + 64 + c0];
        #pragma unroll
        for (int i = 0; i < 8; ++i) {
            const float4 v = src[i];
            const int d = c0 + i * 4;
            float2 p0 = lut_sincos(fmaf(v.x, invk[d + 0], bkk[d + 0]));
            float2 p1 = lut_sincos(fmaf(v.y, invk[d + 1], bkk[d + 1]));
            float2 p2 = lut_sincos(fmaf(v.z, invk[d + 2], bkk[d + 2]));
            float2 p3 = lut_sincos(fmaf(v.w, invk[d + 3], bkk[d + 3]));
            kcos[i * 2 + 0] = pack_bf16(p0.y, p1.y);
            kcos[i * 2 + 1] = pack_bf16(p2.y, p3.y);
            ksin[i * 2 + 0] = pack_bf16(p0.x, p1.x);
            ksin[i * 2 + 1] = pack_bf16(p2.x, p3.x);
        }
    }
    __syncthreads();

    // WMMA fragments per ISA 7.12.2 (wave32, 16-bit):
    //   A 16x32: lanes 0-15 = M; K = hi*8 + e (+8 for e>=8)
    //   B 32x16: same K order as A: K = hi*8 + e (+8 for e>=8) (q replicated in N)
    const int m  = lane & 15;
    const int hi = lane >> 4;

    v8f acc = {};
    #pragma unroll
    for (int kc = 0; kc < 4; ++kc) {
        v16bf a, bf;
        #pragma unroll
        for (int e = 0; e < 16; ++e) {
            int ka = kc * 32 + hi * 8 + e + (e < 8 ? 0 : 8);
            a[e]  = ks[m * 128 + ka];
            bf[e] = qs[ka];
        }
        acc = __builtin_amdgcn_wmma_f32_16x16x32_bf16(
            false, a, false, bf, (short)0, acc, false, false);
    }

    // D: VGPR r -> M=r (lanes 0-15) / M=8+r (lanes 16-31); all N identical.
    if (m == 0) {
        float* dst = scores + ((size_t)(b * 16 + h)) * 4096 + s0 + hi * 8;
        #pragma unroll
        for (int r = 0; r < 8; ++r) dst[r] = acc[r] * INV_SCALE;
    }
}

// ---------------------------------------------------------------------------
// Kernel 2: softmax over S=4096 for one (b,h) row, in place.
// ---------------------------------------------------------------------------
__global__ __launch_bounds__(256)
void euler_softmax_kernel(float* __restrict__ scores)
{
    float* row = scores + (size_t)blockIdx.x * 4096;
    const int tid = threadIdx.x;
    __shared__ float red[256];

    float mx = -INFINITY;
    for (int s = tid; s < 4096; s += 256) mx = fmaxf(mx, row[s]);
    red[tid] = mx; __syncthreads();
    for (int off = 128; off > 0; off >>= 1) {
        if (tid < off) red[tid] = fmaxf(red[tid], red[tid + off]);
        __syncthreads();
    }
    mx = red[0]; __syncthreads();

    float sum = 0.0f;
    for (int s = tid; s < 4096; s += 256) {
        float e = __expf(row[s] - mx);
        row[s] = e;
        sum += e;
    }
    red[tid] = sum; __syncthreads();
    for (int off = 128; off > 0; off >>= 1) {
        if (tid < off) red[tid] += red[tid + off];
        __syncthreads();
    }
    float invs = 1.0f / red[0];
    for (int s = tid; s < 4096; s += 256) row[s] *= invs;
}

// ---------------------------------------------------------------------------
// Kernel 3: heads collapse — combined[b,s] = sum_h weights[b,h,s].
// ---------------------------------------------------------------------------
__global__ __launch_bounds__(256)
void euler_combine_kernel(const float* __restrict__ weights,
                          float* __restrict__ combined)
{
    const int i = blockIdx.x * 256 + threadIdx.x;   // b*4096 + s
    const int b = i >> 12, s = i & 4095;
    float acc = 0.0f;
    #pragma unroll
    for (int h = 0; h < 16; ++h)
        acc += weights[((size_t)(b * 16 + h)) * 4096 + s];
    combined[i] = acc;
}

// ---------------------------------------------------------------------------
// Kernel 4: context partials — second streaming pass over cached_states,
// all global_load_b128. Block = (s-chunk of 128, b); thread t owns columns
// m = 4t..4t+3 of the full 1024-wide row.
// partial[b,sc,m] = sum_{s in chunk} combined[b,s] * cached[b,s,m]
// ---------------------------------------------------------------------------
__global__ __launch_bounds__(256)
void euler_context_kernel(const float* __restrict__ cached,
                          const float* __restrict__ combined,
                          float* __restrict__ partial)
{
    const int sc  = blockIdx.x;   // 0..31 (s chunk of 128)
    const int b   = blockIdx.y;
    const int tid = threadIdx.x;

    __shared__ float wsh[128];
    if (tid < 128) wsh[tid] = combined[b * 4096 + sc * 128 + tid];
    __syncthreads();

    const float4* base = (const float4*)(cached
        + ((size_t)b * 4096 + sc * 128) * 1024) + tid;
    float4 acc = make_float4(0.f, 0.f, 0.f, 0.f);
    #pragma unroll 4
    for (int s = 0; s < 128; ++s) {
        const float  w = wsh[s];
        const float4 v = base[(size_t)s * 256];
        acc.x = fmaf(w, v.x, acc.x);
        acc.y = fmaf(w, v.y, acc.y);
        acc.z = fmaf(w, v.z, acc.z);
        acc.w = fmaf(w, v.w, acc.w);
    }
    ((float4*)(partial + ((size_t)(b * 32 + sc)) * 1024))[tid] = acc;
}

// ---------------------------------------------------------------------------
// Kernel 5: reduce partials and apply the output phase-encode.
// ---------------------------------------------------------------------------
__global__ __launch_bounds__(256)
void euler_final_kernel(const float* __restrict__ partial,
                        const float* __restrict__ t,
                        const float* __restrict__ wo, const float* __restrict__ bo,
                        const float* __restrict__ oscale,
                        float* __restrict__ out)
{
    const int i = blockIdx.x * 256 + threadIdx.x;   // b*1024 + m
    const int b = i >> 10, m = i & 1023;
    float ctx = 0.0f;
    #pragma unroll
    for (int sc = 0; sc < 32; ++sc)
        ctx += partial[((size_t)(b * 32 + sc)) * 1024 + m];
    float inv = 1.0f / (1.0f + fabsf(wo[m]));
    float th  = ctx * inv + bo[m] + t[b] * PHI_F;
    float2 sv = lut_sincos(th);
    out[i] = oscale[m] * (sv.y + sv.x);
}

// ---------------------------------------------------------------------------
extern "C" void kernel_launch(void* const* d_in, const int* in_sizes, int n_in,
                              void* d_out, int out_size, void* d_ws, size_t ws_size,
                              hipStream_t stream)
{
    const float* x      = (const float*)d_in[0];   // [16,1024]
    const float* cached = (const float*)d_in[1];   // [16,4096,1024]
    const float* t      = (const float*)d_in[2];   // [16]
    const float* wq     = (const float*)d_in[3];   // [16,64]
    const float* bq     = (const float*)d_in[4];
    const float* wk     = (const float*)d_in[5];
    const float* bk     = (const float*)d_in[6];
    const float* wo     = (const float*)d_in[7];   // [1024]
    const float* bo     = (const float*)d_in[8];
    const float* osc    = (const float*)d_in[9];
    float* out = (float*)d_out;                    // [16,1024] f32

    float* ws       = (float*)d_ws;
    float* scores   = ws;                          // B*H*S  = 1,048,576 f32
    float* combined = ws + 1048576;                // B*S    =    65,536 f32
    float* partial  = ws + 1048576 + 65536;        // B*32*D =   524,288 f32

    dim3 g1(4096 / 16, 16, 16);                    // (s-tiles, H, B)
    euler_scores_kernel<<<g1, 32, 0, stream>>>(x, cached, t, wq, bq, wk, bk, scores);

    euler_softmax_kernel<<<16 * 16, 256, 0, stream>>>(scores);

    euler_combine_kernel<<<(16 * 4096) / 256, 256, 0, stream>>>(scores, combined);

    dim3 g3(32, 16);                               // (s-chunks, B)
    euler_context_kernel<<<g3, 256, 0, stream>>>(cached, combined, partial);

    euler_final_kernel<<<(16 * 1024) / 256, 256, 0, stream>>>(partial, t, wo, bo, osc, out);
}
